// PolyLoRALinear_24661702213926
// MI455X (gfx1250) — matrix operand order repaired and checked by reading, not compiled, not run on hardware
//
#include <hip/hip_runtime.h>
#include <hip/hip_bf16.h>

// MI455X / gfx1250 (wave32, WMMA, async-to-LDS). Compile-only loop.
//
// out[b,t,o] = x@W^T + bias + (x@A_b)@B_b/16
//   x [8,2048,4096] f32, W [4096,4096] f32  ->  0.55 TFLOP GEMM, compute-bound.
//
// Path A (ws >= ~173MB): pre-convert x/W to bf16 in ws, then a double-buffered
//   128x256x64 WMMA GEMM staged entirely with global_load_async_to_lds_b128
//   (ASYNCcnt, s_wait_asynccnt partial waits). Rank-16 adapter folded into the
//   accumulators with one extra WMMA k-step (K=16 zero-padded to 32).
// Path B (small ws): 128x128 GEMM with in-kernel f32->bf16 staging.
//
// NOTE: reference task_ids is jnp.int64; per harness contract we read int32.

typedef __attribute__((ext_vector_type(16))) __bf16 v16bf;
typedef __attribute__((ext_vector_type(8)))  __bf16 v8bf;
typedef __attribute__((ext_vector_type(8)))  float  v8f;

#define LDT 72   // GEMM tile row stride (ushorts): 64 data + 8 pad -> 144B (16B aligned)
#define LDU 40   // adapter tile row stride (ushorts): 32 data + 8 pad -> 80B (16B aligned)

__device__ __forceinline__ unsigned short f2bf(float f) {
    unsigned int u = __float_as_uint(f);
    u += 0x7FFFu + ((u >> 16) & 1u);   // round-to-nearest-even
    return (unsigned short)(u >> 16);
}
__device__ __forceinline__ unsigned pk2(float a, float b) {
    return (unsigned)f2bf(a) | ((unsigned)f2bf(b) << 16);
}

// A fragment (16-bit A layout): lane holds chunk [base..base+7] and [base+16..base+23].
__device__ __forceinline__ v16bf ld_afrag(const unsigned short* base) {
    v8bf lo = *(const v8bf*)base;
    v8bf hi = *(const v8bf*)(base + 16);
    return __builtin_shufflevector(lo, hi, 0,1,2,3,4,5,6,7,8,9,10,11,12,13,14,15);
}
// B fragment: lane holds chunk [base..base+7] and [base+16..base+23] (same K order as A).
__device__ __forceinline__ v16bf ld_bfrag(const unsigned short* base) {
    v8bf lo = *(const v8bf*)base;
    v8bf hi = *(const v8bf*)(base + 16);
    return __builtin_shufflevector(lo, hi, 0,1,2,3,4,5,6,7,8,9,10,11,12,13,14,15);
}

// gfx1250 async memory->LDS, tracked by ASYNCcnt (inline asm: portable across
// ROCm/upstream toolchains whose builtin arities differ).
__device__ __forceinline__ void async_ld_b128(const void* gptr, const void* lptr) {
    unsigned lds = (unsigned)(uintptr_t)lptr;                 // low 32 bits = LDS offset
    unsigned long long ga = (unsigned long long)(uintptr_t)gptr;
    asm volatile("global_load_async_to_lds_b128 %0, %1, off"
                 :: "v"(lds), "v"(ga) : "memory");
}
#define ASYNC_WAIT_12() asm volatile("s_wait_asynccnt 12" ::: "memory")
#define ASYNC_WAIT_0()  asm volatile("s_wait_asynccnt 0"  ::: "memory")

// ---------------------------------------------------------------------------
// 1) selector: mw[b,s] = sigmoid(logits[task[b],s]) / (sum_s + eps)
// ---------------------------------------------------------------------------
__global__ void poly_sel_kernel(const float* __restrict__ logits,
                                const int* __restrict__ task_ids,
                                float* __restrict__ mw) {
    int b = threadIdx.x;
    if (b < 8) {
        int t = task_ids[b];
        float v[8];
        float s = 0.f;
#pragma unroll
        for (int i = 0; i < 8; ++i) {
            v[i] = 1.f / (1.f + __expf(-logits[t * 8 + i]));
            s += v[i];
        }
        float inv = 1.f / (s + 1e-12f);
#pragma unroll
        for (int i = 0; i < 8; ++i) mw[b * 8 + i] = v[i] * inv;
    }
}

// ---------------------------------------------------------------------------
// 2a) Amix[b,d,r] = sum_s mw[b,s] * lora_a[s,d,r]
// ---------------------------------------------------------------------------
__global__ void poly_mixA_kernel(const float* __restrict__ mw,
                                 const float* __restrict__ lora_a,
                                 float* __restrict__ Amix) {
    int gid = blockIdx.x * 256 + threadIdx.x;    // 8*4096 threads
    int b = gid >> 12;
    int d = gid & 4095;
    float w[8];
#pragma unroll
    for (int s = 0; s < 8; ++s) w[s] = mw[b * 8 + s];
    float acc[16];
#pragma unroll
    for (int r = 0; r < 16; ++r) acc[r] = 0.f;
#pragma unroll
    for (int s = 0; s < 8; ++s) {
        const float* p = lora_a + ((size_t)s * 4096 + d) * 16;
#pragma unroll
        for (int r = 0; r < 16; ++r) acc[r] = fmaf(w[s], p[r], acc[r]);
    }
    float* o = Amix + ((size_t)b * 4096 + d) * 16;
#pragma unroll
    for (int r = 0; r < 16; ++r) o[r] = acc[r];
}

// ---------------------------------------------------------------------------
// 2b) Bmix[b,r,o] = (1/rank) * sum_s mw[b,s] * lora_b[s,r,o]
// ---------------------------------------------------------------------------
__global__ void poly_mixB_kernel(const float* __restrict__ mw,
                                 const float* __restrict__ lora_b,
                                 float* __restrict__ Bmix) {
    int gid = blockIdx.x * 256 + threadIdx.x;    // 8*16*4096 threads
    int o = gid & 4095;
    int r = (gid >> 12) & 15;
    int b = gid >> 16;
    float a = 0.f;
#pragma unroll
    for (int s = 0; s < 8; ++s)
        a = fmaf(mw[b * 8 + s], lora_b[((size_t)s * 16 + r) * 4096 + o], a);
    Bmix[gid] = a * (1.f / 16.f);
}

// ---------------------------------------------------------------------------
// 3) u[m,r] = sum_k x[m,k] * Amix[b,k,r]   (one wave32 per row m)
// ---------------------------------------------------------------------------
__global__ void poly_xu_kernel(const float* __restrict__ x,
                               const float* __restrict__ Amix,
                               float* __restrict__ u) {
    int lane = threadIdx.x & 31;
    int wid  = threadIdx.x >> 5;
    int m = blockIdx.x * 8 + wid;
    int b = m >> 11;
    const float* xr = x + (size_t)m * 4096;
    const float* Ab = Amix + (size_t)b * 4096 * 16;
    float acc[16];
#pragma unroll
    for (int r = 0; r < 16; ++r) acc[r] = 0.f;
    for (int k = lane; k < 4096; k += 32) {
        float xv = xr[k];
        const float* ar = Ab + (size_t)k * 16;
#pragma unroll
        for (int r = 0; r < 16; ++r) acc[r] = fmaf(xv, ar[r], acc[r]);
    }
#pragma unroll
    for (int off = 16; off; off >>= 1) {
#pragma unroll
        for (int r = 0; r < 16; ++r) acc[r] += __shfl_xor(acc[r], off, 32);
    }
    if (lane == 0) {
        float* o = u + (size_t)m * 16;
#pragma unroll
        for (int r = 0; r < 16; ++r) o[r] = acc[r];
    }
}

// ---------------------------------------------------------------------------
// 4) one-shot f32 -> bf16 pre-convert (8 elements / thread, 16B stores)
// ---------------------------------------------------------------------------
__global__ void poly_cvt_bf16_kernel(const float* __restrict__ in,
                                     unsigned short* __restrict__ out) {
    size_t i = ((size_t)blockIdx.x * 256 + threadIdx.x) * 8;
    const float4* p = (const float4*)(in + i);
    float4 a0 = p[0], a1 = p[1];
    uint4 d;
    d.x = pk2(a0.x, a0.y);
    d.y = pk2(a0.z, a0.w);
    d.z = pk2(a1.x, a1.y);
    d.w = pk2(a1.z, a1.w);
    *(uint4*)(out + i) = d;
}

// ---------------------------------------------------------------------------
// 5) PATH A: 128x256x64 bf16 WMMA GEMM, async-to-LDS double-buffered staging,
//    WMMA-folded rank-16 adapter + bias epilogue.
// ---------------------------------------------------------------------------
#define A_BM 128
#define A_BN 256
#define A_BK 64
#define SX_BYTES (A_BM * LDT * 2)       // 18432
#define SW_BYTES (A_BN * LDT * 2)       // 36864

__device__ __forceinline__ void issue_tile_x(const unsigned short* g,
                                             unsigned short* s, int tid) {
#pragma unroll
    for (int i = 0; i < 4; ++i) {                  // 1024 chunks of 8 bf16
        int c = tid + 256 * i;
        int row = c >> 3;
        int col = (c & 7) * 8;
        async_ld_b128(g + (size_t)row * 4096 + col, s + row * LDT + col);
    }
}
__device__ __forceinline__ void issue_tile_w(const unsigned short* g,
                                             unsigned short* s, int tid) {
#pragma unroll
    for (int i = 0; i < 8; ++i) {                  // 2048 chunks of 8 bf16
        int c = tid + 256 * i;
        int row = c >> 3;
        int col = (c & 7) * 8;
        async_ld_b128(g + (size_t)row * 4096 + col, s + row * LDT + col);
    }
}

__global__ __launch_bounds__(256)
void poly_gemm_bf16_kernel(const unsigned short* __restrict__ xb,
                           const unsigned short* __restrict__ wb,
                           const float* __restrict__ bias,
                           const float* __restrict__ u,
                           const float* __restrict__ Bmix,
                           float* __restrict__ out) {
    // Carved LDS: [sX0 | sX1 | sW0 | sW1]; epilogue tiles alias the X buffers.
    // (No local pointer arrays over LDS: ld.lld rejects addrspacecast in
    // static initializers — all buffer pointers are runtime byte-offset math.)
    __shared__ __align__(16) unsigned char smem[2 * SX_BYTES + 2 * SW_BYTES]; // 110592B
    unsigned char* smp = smem;

    const int tid   = threadIdx.x;
    const int lane  = tid & 31;
    const int wid   = tid >> 5;
    const int waveM = wid & 3;      // 4 waves x 32 rows
    const int waveN = wid >> 2;     // 2 waves x 128 cols
    const int half  = lane >> 4;
    const int l16   = lane & 15;

    const int blockN = blockIdx.x * A_BN;
    const int blockM = blockIdx.y * A_BM;
    const int b      = blockM >> 11;

    const unsigned short* gx = xb + (size_t)blockM * 4096;
    const unsigned short* gw = wb + (size_t)blockN * 4096;

    v8f acc[2][8];
#pragma unroll
    for (int mt = 0; mt < 2; ++mt)
#pragma unroll
        for (int nt = 0; nt < 8; ++nt) acc[mt][nt] = 0;

    // Prologue: prefetch tiles 0 and 1 (12 async instrs / wave / tile).
    issue_tile_x(gx + 0 * A_BK, (unsigned short*)(smp), tid);
    issue_tile_w(gw + 0 * A_BK, (unsigned short*)(smp + 2 * SX_BYTES), tid);
    issue_tile_x(gx + 1 * A_BK, (unsigned short*)(smp + SX_BYTES), tid);
    issue_tile_w(gw + 1 * A_BK, (unsigned short*)(smp + 2 * SX_BYTES + SW_BYTES), tid);
    ASYNC_WAIT_12();            // in-order completion: tile0 resident
    __syncthreads();

    const int NT = 4096 / A_BK; // 64
    for (int t = 0; t < NT; ++t) {
        const int cur = t & 1;
        unsigned short* sXc = (unsigned short*)(smp + cur * SX_BYTES);
        unsigned short* sWc = (unsigned short*)(smp + 2 * SX_BYTES + cur * SW_BYTES);

#pragma unroll
        for (int kk = 0; kk < A_BK; kk += 32) {
            v16bf af0 = ld_afrag(sXc + (waveM * 32 + l16) * LDT + kk + half * 8);
            v16bf af1 = ld_afrag(sXc + (waveM * 32 + 16 + l16) * LDT + kk + half * 8);
            // Register-rotated B fragments: load nt+1 before the WMMAs of nt so
            // the ds_load latency hides behind two matrix ops.
            v16bf bcur = ld_bfrag(sWc + (waveN * 128 + l16) * LDT + kk + half * 8);
#pragma unroll
            for (int nt = 0; nt < 8; ++nt) {
                v16bf bnxt = bcur;
                if (nt < 7)
                    bnxt = ld_bfrag(sWc + (waveN * 128 + (nt + 1) * 16 + l16) * LDT
                                    + kk + half * 8);
                acc[0][nt] = __builtin_amdgcn_wmma_f32_16x16x32_bf16(
                    false, af0, false, bcur, (short)0, acc[0][nt], false, false);
                acc[1][nt] = __builtin_amdgcn_wmma_f32_16x16x32_bf16(
                    false, af1, false, bcur, (short)0, acc[1][nt], false, false);
                bcur = bnxt;
            }
        }
        __syncthreads();                       // all waves done reading buf cur
        if (t + 2 < NT) {
            issue_tile_x(gx + (size_t)(t + 2) * A_BK, sXc, tid);
            issue_tile_w(gw + (size_t)(t + 2) * A_BK, sWc, tid);
            ASYNC_WAIT_12();                   // my tile t+1 loads complete
        } else {
            ASYNC_WAIT_0();
        }
        __syncthreads();                       // everyone's tile t+1 resident
    }

    // ---- adapter as one extra WMMA k-step: acc += u[128x16] @ Bmix[16x256] ----
    // Stage u (A layout, K 16->32 zero-padded) and Bmix^T (B layout: row = N,
    // cols = r zero-padded) as bf16 in the retired X buffers.
    unsigned short* sUb = (unsigned short*)smp;            // 128*LDU*2 = 10240B
    unsigned short* sBt = (unsigned short*)(smp + 10240);  // 256*LDU*2 = 20480B (<36864)
#pragma unroll
    for (int i = 0; i < 2; ++i) {              // 512 chunks of 8
        int c = tid + 256 * i;
        int row = c >> 2;
        int cc = c & 3;
        uint4 d = {0u, 0u, 0u, 0u};
        if (cc < 2) {
            const float* ur = u + ((size_t)blockM + row) * 16 + cc * 8;
            d.x = pk2(ur[0], ur[1]); d.y = pk2(ur[2], ur[3]);
            d.z = pk2(ur[4], ur[5]); d.w = pk2(ur[6], ur[7]);
        }
        *(uint4*)(sUb + row * LDU + cc * 8) = d;
    }
#pragma unroll
    for (int i = 0; i < 4; ++i) {              // 1024 chunks of 8
        int c = tid + 256 * i;
        int n = c >> 2;
        int cc = c & 3;
        uint4 d = {0u, 0u, 0u, 0u};
        if (cc < 2) {
            const float* bp = Bmix + (size_t)b * 16 * 4096
                            + (size_t)(cc * 8) * 4096 + blockN + n;
            d.x = pk2(bp[0], bp[4096]);
            d.y = pk2(bp[2 * 4096], bp[3 * 4096]);
            d.z = pk2(bp[4 * 4096], bp[5 * 4096]);
            d.w = pk2(bp[6 * 4096], bp[7 * 4096]);
        }
        *(uint4*)(sBt + n * LDU + cc * 8) = d;
    }
    __syncthreads();

    {
        v16bf ua0 = ld_afrag(sUb + (waveM * 32 + l16) * LDU + half * 8);
        v16bf ua1 = ld_afrag(sUb + (waveM * 32 + 16 + l16) * LDU + half * 8);
#pragma unroll
        for (int nt = 0; nt < 8; ++nt) {
            v16bf bfrag = ld_bfrag(sBt + (waveN * 128 + nt * 16 + l16) * LDU + half * 8);
            acc[0][nt] = __builtin_amdgcn_wmma_f32_16x16x32_bf16(
                false, ua0, false, bfrag, (short)0, acc[0][nt], false, false);
            acc[1][nt] = __builtin_amdgcn_wmma_f32_16x16x32_bf16(
                false, ua1, false, bfrag, (short)0, acc[1][nt], false, false);
        }
    }

    // ---- store: + bias. C layout: VGPR v, lanes 0-15 -> M=v; lanes 16-31 -> M=8+v.
#pragma unroll
    for (int mt = 0; mt < 2; ++mt) {
#pragma unroll
        for (int v = 0; v < 8; ++v) {
            int mloc = waveM * 32 + mt * 16 + half * 8 + v;
            size_t m = (size_t)blockM + mloc;
#pragma unroll
            for (int nt = 0; nt < 8; ++nt) {
                int nloc = waveN * 128 + nt * 16 + l16;
                out[m * 4096 + blockN + nloc] = acc[mt][nt][v] + bias[blockN + nloc];
            }
        }
    }
}

// ---------------------------------------------------------------------------
// 6) PATH B fallback: 128x128 GEMM with in-kernel f32->bf16 staging.
// ---------------------------------------------------------------------------
#define B_BM 128
#define B_BN 128
#define B_BK 64

__global__ __launch_bounds__(256)
void poly_gemm_f32_kernel(const float* __restrict__ x,
                          const float* __restrict__ w,
                          const float* __restrict__ bias,
                          const float* __restrict__ u,
                          const float* __restrict__ Bmix,
                          float* __restrict__ out) {
    __shared__ __align__(16) unsigned short sX[B_BM * LDT];
    __shared__ __align__(16) unsigned short sW[B_BN * LDT];
    __shared__ float sU[B_BM * 16];
    __shared__ float sB[16 * B_BN];

    const int tid   = threadIdx.x;
    const int lane  = tid & 31;
    const int wid   = tid >> 5;
    const int waveM = wid & 3;
    const int waveN = wid >> 2;
    const int half  = lane >> 4;
    const int l16   = lane & 15;

    const int blockN = blockIdx.x * B_BN;
    const int blockM = blockIdx.y * B_BM;
    const int b      = blockM >> 11;

#pragma unroll
    for (int i = 0; i < 8; ++i) {
        int idx = tid + 256 * i;
        sU[idx] = u[(size_t)blockM * 16 + idx];
        int r = idx >> 7, n = idx & 127;
        sB[idx] = Bmix[(size_t)b * 16 * 4096 + (size_t)r * 4096 + blockN + n];
    }

    v8f acc[2][4];
#pragma unroll
    for (int mt = 0; mt < 2; ++mt)
#pragma unroll
        for (int nt = 0; nt < 4; ++nt) acc[mt][nt] = 0;

    const float* gx = x + (size_t)blockM * 4096;
    const float* gw = w + (size_t)blockN * 4096;

    for (int k0 = 0; k0 < 4096; k0 += B_BK) {
#pragma unroll
        for (int i = 0; i < 4; ++i) {
            int c   = tid + 256 * i;
            int row = c >> 3;
            int col = (c & 7) * 8;
            {
                const float4* p = (const float4*)(gx + (size_t)row * 4096 + k0 + col);
                float4 a0 = p[0], a1 = p[1];
                uint4 d;
                d.x = pk2(a0.x, a0.y); d.y = pk2(a0.z, a0.w);
                d.z = pk2(a1.x, a1.y); d.w = pk2(a1.z, a1.w);
                *(uint4*)(sX + row * LDT + col) = d;
            }
            {
                const float4* p = (const float4*)(gw + (size_t)row * 4096 + k0 + col);
                float4 a0 = p[0], a1 = p[1];
                uint4 d;
                d.x = pk2(a0.x, a0.y); d.y = pk2(a0.z, a0.w);
                d.z = pk2(a1.x, a1.y); d.w = pk2(a1.z, a1.w);
                *(uint4*)(sW + row * LDT + col) = d;
            }
        }
        __syncthreads();

#pragma unroll
        for (int kk = 0; kk < B_BK; kk += 32) {
            v16bf af0 = ld_afrag(sX + (waveM * 32 + l16) * LDT + kk + half * 8);
            v16bf af1 = ld_afrag(sX + (waveM * 32 + 16 + l16) * LDT + kk + half * 8);
            v16bf bcur = ld_bfrag(sW + (waveN * 64 + l16) * LDT + kk + half * 8);
#pragma unroll
            for (int nt = 0; nt < 4; ++nt) {
                v16bf bnxt = bcur;
                if (nt < 3)
                    bnxt = ld_bfrag(sW + (waveN * 64 + (nt + 1) * 16 + l16) * LDT
                                    + kk + half * 8);
                acc[0][nt] = __builtin_amdgcn_wmma_f32_16x16x32_bf16(
                    false, af0, false, bcur, (short)0, acc[0][nt], false, false);
                acc[1][nt] = __builtin_amdgcn_wmma_f32_16x16x32_bf16(
                    false, af1, false, bcur, (short)0, acc[1][nt], false, false);
                bcur = bnxt;
            }
        }
        __syncthreads();
    }

#pragma unroll
    for (int mt = 0; mt < 2; ++mt) {
#pragma unroll
        for (int v = 0; v < 8; ++v) {
            int mloc = waveM * 32 + mt * 16 + half * 8 + v;
            size_t m = (size_t)blockM + mloc;
            float uu[16];
#pragma unroll
            for (int r = 0; r < 16; ++r) uu[r] = sU[mloc * 16 + r];
#pragma unroll
            for (int nt = 0; nt < 4; ++nt) {
                int nloc = waveN * 64 + nt * 16 + l16;
                float adapt = 0.f;
#pragma unroll
                for (int r = 0; r < 16; ++r)
                    adapt = fmaf(uu[r], sB[r * 128 + nloc], adapt);
                out[m * 4096 + blockN + nloc] =
                    acc[mt][nt][v] + bias[blockN + nloc] + adapt;
            }
        }
    }
}

// ---------------------------------------------------------------------------
extern "C" void kernel_launch(void* const* d_in, const int* in_sizes, int n_in,
                              void* d_out, int out_size, void* d_ws, size_t ws_size,
                              hipStream_t stream) {
    (void)in_sizes; (void)n_in; (void)out_size;

    const float* x       = (const float*)d_in[0];   // [8,2048,4096]
    const float* weight  = (const float*)d_in[1];   // [4096,4096]
    const float* bias    = (const float*)d_in[2];   // [4096]
    const float* logits  = (const float*)d_in[3];   // [64,8]
    const float* lora_a  = (const float*)d_in[4];   // [1,8,4096,16]
    const float* lora_b  = (const float*)d_in[5];   // [1,8,16,4096]
    const int*   task_id = (const int*)d_in[6];     // [8]
    float* out = (float*)d_out;                     // [8,2048,4096]

    const size_t NX = (size_t)8 * 2048 * 4096;      // 67,108,864
    const size_t NW = (size_t)4096 * 4096;          // 16,777,216

    float* ws   = (float*)d_ws;
    float* mw   = ws;                               // 64
    float* Amix = ws + 64;                          // 8*4096*16
    float* Bmix = Amix + (size_t)8 * 4096 * 16;     // 8*16*4096
    float* u    = Bmix + (size_t)8 * 16 * 4096;     // 16384*16
    const size_t f32_ws_bytes = (64 + (size_t)8*4096*16 + (size_t)8*16*4096
                                 + (size_t)16384*16) * sizeof(float);
    unsigned short* xb = (unsigned short*)((char*)d_ws + ((f32_ws_bytes + 255) & ~(size_t)255));
    unsigned short* wb = xb + NX;
    const size_t need = ((f32_ws_bytes + 255) & ~(size_t)255) + (NX + NW) * 2;

    poly_sel_kernel<<<1, 32, 0, stream>>>(logits, task_id, mw);
    poly_mixA_kernel<<<(8 * 4096) / 256, 256, 0, stream>>>(mw, lora_a, Amix);
    poly_mixB_kernel<<<(8 * 16 * 4096) / 256, 256, 0, stream>>>(mw, lora_b, Bmix);
    poly_xu_kernel<<<16384 / 8, 256, 0, stream>>>(x, Amix, u);

    if (ws_size >= need) {
        poly_cvt_bf16_kernel<<<(int)(NX / 2048), 256, 0, stream>>>(x, xb);
        poly_cvt_bf16_kernel<<<(int)(NW / 2048), 256, 0, stream>>>(weight, wb);
        poly_gemm_bf16_kernel<<<dim3(4096 / A_BN, 16384 / A_BM), 256, 0, stream>>>(
            xb, wb, bias, u, Bmix, out);
    } else {
        poly_gemm_f32_kernel<<<dim3(4096 / B_BN, 16384 / B_BM), 256, 0, stream>>>(
            x, weight, bias, u, Bmix, out);
    }
}
